// Cross_Attention_Sparse_84456236909403
// MI455X (gfx1250) — matrix operand order repaired and checked by reading, not compiled, not run on hardware
//
#include <hip/hip_runtime.h>

// ---------------------------------------------------------------------------
// Cross-attention with bidirectional top-k sparsity, CDNA5 / gfx1250 (wave32).
// All matmuls run on v_wmma_f32_16x16x32_bf16. S (bf16, 128MB) is staged in
// workspace so it can live in the 192MB L2 across the select/softmax passes.
// V tiles for the P@V pass are staged into LDS by the Tensor Data Mover
// (tensor_load_to_lds + s_wait_tensorcnt). Workspace need: ~148.4 MB.
// ---------------------------------------------------------------------------

typedef __attribute__((ext_vector_type(16))) __bf16   v16bf;
typedef __attribute__((ext_vector_type(8)))  __bf16   v8bf;
typedef __attribute__((ext_vector_type(8)))  float    v8f;
typedef __attribute__((ext_vector_type(4)))  unsigned v4u;
typedef __attribute__((ext_vector_type(8)))  unsigned v8u;

#define B_    2
#define N_    2048
#define C_    512
#define H_    8
#define HD_   64
#define BH_   (B_ * H_)
#define KTOP_ 1024
#define SCALE_ 0.125f   // HD^-0.5 = 1/8

// ---- bf16 <-> f32 helpers (bit-level, round-to-nearest-even) ---------------
static __device__ __forceinline__ __bf16 f2bf(float f) {
  unsigned u = __float_as_uint(f);
  unsigned r = u + 0x7fffu + ((u >> 16) & 1u);
  unsigned short h = (unsigned short)(r >> 16);
  return __builtin_bit_cast(__bf16, h);
}
static __device__ __forceinline__ float bf2f(__bf16 b) {
  unsigned short h = __builtin_bit_cast(unsigned short, b);
  return __uint_as_float(((unsigned)h) << 16);
}
// order-preserving float->uint key (descending select works on raw compares)
static __device__ __forceinline__ unsigned okey(float x) {
  unsigned u = __float_as_uint(x);
  return (u & 0x80000000u) ? ~u : (u | 0x80000000u);
}
static __device__ __forceinline__ float oinv(unsigned k) {
  unsigned u = (k & 0x80000000u) ? (k & 0x7fffffffu) : ~k;
  return __uint_as_float(u);
}

// ---- WMMA fragment loaders (bf16 16x16x32, per ISA 7.12.2 layouts) ---------
// A fragment (16xK tile, row-major, row stride ld):
//   lane -> row = lane&15 ; half j -> K = (j>>3)*16 + (lane>>4)*8 + (j&7)
static __device__ __forceinline__ v16bf ldfragA(const __bf16* __restrict__ base,
                                                int lane, int ld) {
  const __bf16* p = base + (lane & 15) * ld + ((lane >> 4) << 3);
  v8bf lo = *(const v8bf*)(p);
  v8bf hi = *(const v8bf*)(p + 16);
  v16bf a;
#pragma unroll
  for (int i = 0; i < 8; ++i) { a[i] = lo[i]; a[i + 8] = hi[i]; }
  return a;
}
// B fragment from a [N][K] tile (i.e. B^T rows; row stride ld):
//   lane -> col N = lane&15 ; half j -> K = (j>>3)*16 + (lane>>4)*8 + (j&7)
static __device__ __forceinline__ v16bf ldfragB(const __bf16* __restrict__ base,
                                                int lane, int ld) {
  const __bf16* p = base + (lane & 15) * ld + ((lane >> 4) << 3);
  v8bf lo = *(const v8bf*)(p);
  v8bf hi = *(const v8bf*)(p + 16);
  v16bf b;
#pragma unroll
  for (int i = 0; i < 8; ++i) { b[i] = lo[i]; b[i + 8] = hi[i]; }
  return b;
}
// B fragment gathered from a natural [K][C] tile (row stride ldc), as left by
// the TDM: element (K = (j>>3)*16 + (lane>>4)*8 + (j&7), N = lane&15).
static __device__ __forceinline__ v16bf ldfragB_kc(const __bf16* __restrict__ base,
                                                   int lane, int ldc) {
  const __bf16* p = base + (lane & 15) + (((lane >> 4) << 3) * ldc);
  v16bf b;
#pragma unroll
  for (int j = 0; j < 16; ++j) b[j] = p[(j + (j & 8)) * ldc];
  return b;
}
static __device__ __forceinline__ v8f wmma_bf16(v16bf a, v16bf b, v8f c) {
  return __builtin_amdgcn_wmma_f32_16x16x32_bf16(false, a, false, b,
                                                 (short)0, c, false, false);
}

// ---------------------------------------------------------------------------
// Kernel 1: Y = X @ W^T, output head-split bf16 [B,H,N,HD].
// Block = 64x64 output tile, 8 waves, 2 subtiles/wave. LDS-staged bf16 tiles.
// ---------------------------------------------------------------------------
__global__ __launch_bounds__(256) void proj_kernel(const float* __restrict__ X,
                                                   const float* __restrict__ W,
                                                   __bf16* __restrict__ Y) {
  __shared__ __bf16 Xs[64 * 32];
  __shared__ __bf16 Ws[64 * 32];
  const int tid = threadIdx.x, lane = tid & 31, w = tid >> 5;
  const int m0 = blockIdx.y * 64;
  const int c0 = blockIdx.x * 64;
  const int sr = w >> 1;         // subtile row group 0..3
  const int sc = (w & 1) * 2;    // subtile col group 0 / 2

  v8f acc0 = {}, acc1 = {};
  for (int kk = 0; kk < C_; kk += 32) {
    for (int i = tid; i < 64 * 32; i += 256) {
      int r = i >> 5, c = i & 31;
      Xs[i] = f2bf(X[(size_t)(m0 + r) * C_ + kk + c]);
      Ws[i] = f2bf(W[(size_t)(c0 + r) * C_ + kk + c]);   // W rows == (W^T)^T
    }
    __syncthreads();
    v16bf a  = ldfragA(Xs + sr * 16 * 32, lane, 32);
    v16bf b0 = ldfragB(Ws + (sc * 16) * 32, lane, 32);
    v16bf b1 = ldfragB(Ws + ((sc + 1) * 16) * 32, lane, 32);
    acc0 = wmma_bf16(a, b0, acc0);
    acc1 = wmma_bf16(a, b1, acc1);
    __syncthreads();
  }
#pragma unroll
  for (int r = 0; r < 8; ++r) {
    int m = m0 + sr * 16 + r + ((lane >> 4) << 3);
    int b = m >> 11, n = m & (N_ - 1);
#pragma unroll
    for (int j = 0; j < 2; ++j) {
      int c = c0 + (sc + j) * 16 + (lane & 15);
      int h = c >> 6, d = c & 63;
      float v = (j == 0) ? acc0[r] : acc1[r];
      Y[(((size_t)(b * H_ + h) * N_ + n) << 6) + d] = f2bf(v);
    }
  }
}

// ---------------------------------------------------------------------------
// Kernel 2: S = SCALE * Qh @ Kh^T per (b,h). Block = 128x128 tile of S.
// Fragments loaded directly from global (rows are K-contiguous, HD=64).
// ---------------------------------------------------------------------------
__global__ __launch_bounds__(256) void scores_kernel(const __bf16* __restrict__ Qh,
                                                     const __bf16* __restrict__ Kh,
                                                     __bf16* __restrict__ S) {
  const int tid = threadIdx.x, lane = tid & 31, w = tid >> 5;
  const int bh = blockIdx.z;
  const __bf16* qb = Qh + (size_t)bh * N_ * HD_;
  const __bf16* kb = Kh + (size_t)bh * N_ * HD_;
  __bf16* sb = S + (size_t)bh * N_ * N_;
  const int r0 = blockIdx.y * 128 + (w >> 1) * 32;
  const int c0 = blockIdx.x * 128 + (w & 1) * 64;

  v8f acc[2][4] = {};
#pragma unroll
  for (int kk = 0; kk < HD_; kk += 32) {
    v16bf a0 = ldfragA(qb + (size_t)r0 * HD_ + kk, lane, HD_);
    v16bf a1 = ldfragA(qb + (size_t)(r0 + 16) * HD_ + kk, lane, HD_);
#pragma unroll
    for (int j = 0; j < 4; ++j) {
      v16bf b = ldfragB(kb + (size_t)(c0 + j * 16) * HD_ + kk, lane, HD_);
      acc[0][j] = wmma_bf16(a0, b, acc[0][j]);
      acc[1][j] = wmma_bf16(a1, b, acc[1][j]);
    }
  }
#pragma unroll
  for (int i = 0; i < 2; ++i)
#pragma unroll
    for (int j = 0; j < 4; ++j)
#pragma unroll
      for (int r = 0; r < 8; ++r) {
        int row = r0 + i * 16 + r + ((lane >> 4) << 3);
        int col = c0 + j * 16 + (lane & 15);
        sb[(size_t)row * N_ + col] = f2bf(acc[i][j][r] * SCALE_);
      }
}

// ---------------------------------------------------------------------------
// Kernel 3: exact k-th-largest per row (colMode=0) or per column (colMode=1)
// via 4-pass 8-bit radix select on order-preserving keys. Also row max.
// One 256-thread block per row/column (2048 elements, 8 per thread).
// ---------------------------------------------------------------------------
__global__ __launch_bounds__(256) void select_kernel(const __bf16* __restrict__ S,
                                                     float* __restrict__ t_out,
                                                     float* __restrict__ m_out,
                                                     int colMode) {
  __shared__ unsigned hist[256];
  __shared__ unsigned s_bin, s_above, s_maxkey;
  const int tid = threadIdx.x;
  const int rid = blockIdx.x;         // [0, BH*N)
  const int bh = rid >> 11;
  const int r  = rid & (N_ - 1);
  const __bf16* base = S + (size_t)bh * N_ * N_;

  unsigned prefix = 0, pmask = 0;
  int remaining = KTOP_;
  if (tid == 0) s_maxkey = 0;

  for (int pass = 3; pass >= 0; --pass) {
    hist[tid] = 0;
    __syncthreads();
    const int shift = pass << 3;
    unsigned lmax = 0;
#pragma unroll
    for (int e = 0; e < N_ / 256; ++e) {
      int i = e * 256 + tid;
      float x = bf2f(colMode ? base[(size_t)i * N_ + r]
                             : base[(size_t)r * N_ + i]);
      unsigned key = okey(x);
      if (pass == 3) lmax = key > lmax ? key : lmax;
      if ((key & pmask) == prefix) atomicAdd(&hist[(key >> shift) & 255u], 1u);
    }
    if (pass == 3) atomicMax(&s_maxkey, lmax);
    __syncthreads();
    if (tid == 0) {
      unsigned acc = 0;
      int b = 255;
      for (; b > 0; --b) {
        unsigned c = hist[b];
        if (acc + c >= (unsigned)remaining) break;
        acc += c;
      }
      s_bin = (unsigned)b;
      s_above = acc;
    }
    __syncthreads();
    remaining -= (int)s_above;
    prefix |= s_bin << shift;
    pmask  |= 255u << shift;
  }
  if (tid == 0) {
    t_out[rid] = oinv(prefix);
    if (m_out) m_out[rid] = oinv(s_maxkey);
  }
}

// ---------------------------------------------------------------------------
// Kernel 4: masked softmax + P@V per (b,h). Each wave owns 16 rows; the block
// shares a TDM-staged V tile in LDS (natural [k][c] layout). P is rebuilt on
// the fly as a bf16 A-fragment.
// ---------------------------------------------------------------------------
__global__ __launch_bounds__(256) void softmax_av_kernel(const __bf16* __restrict__ S,
                                                         const __bf16* __restrict__ Vh,
                                                         const float* __restrict__ t_row,
                                                         const float* __restrict__ t_col,
                                                         const float* __restrict__ m_row,
                                                         float* __restrict__ O) {
  __shared__ __bf16 Vs[32 * HD_];    // V chunk, natural [k][c], ld = 64
  __shared__ float tcs[32];
  const int tid = threadIdx.x, lane = tid & 31, w = tid >> 5;
  const int bh = blockIdx.x >> 4;
  const int n0 = (blockIdx.x & 15) * 128 + w * 16;
  const __bf16* sb = S + (size_t)bh * N_ * N_;
  const __bf16* vb = Vh + (size_t)bh * N_ * HD_;
  const int gr = n0 + (lane & 15);
  const float trow = t_row[bh * N_ + gr];
  const float mrow = m_row[bh * N_ + gr];
  const int kb = (lane >> 4) << 3;
  // LDS byte offset of Vs: low 32 bits of the generic (flat) address.
  const unsigned ldsVs = (unsigned)(size_t)(&Vs[0]);

  v8f acc[4] = {};
  float rowsum = 0.f;

  for (int kk = 0; kk < N_; kk += 32) {
    if (w == 0) {
      // ---- Tensor Data Mover: 2D tile [32 rows x 64 elems] bf16 -> LDS ----
      unsigned long long ga = (unsigned long long)(size_t)(vb + (size_t)kk * HD_);
      v4u g0 = { 1u,                                   // count=1 (valid D#)
                 ldsVs,                                // lds_addr (bytes)
                 (unsigned)ga,                         // global_addr[31:0]
                 (unsigned)((ga >> 32) & 0x01ffffffu) | (2u << 30) }; // type=2
      v8u g1 = { 0x00010000u,          // data_size=2B, mask/flags=0
                 (unsigned)(HD_ << 16),// tensor_dim0[15:0]=64 @ bits63:48
                 (unsigned)(HD_ >> 16) | (unsigned)(N_ << 16), // dim0 hi, tensor_dim1=2048
                 (unsigned)(HD_ << 16),// tensor_dim1 hi=0, tile_dim0=64 @ bits127:112
                 32u,                  // tile_dim1=32, tile_dim2=0
                 (unsigned)HD_,        // tensor_dim0_stride=64 (low 32)
                 0u, 0u };             // stride hi, tensor_dim1_stride=0
      v4u gz = { 0u, 0u, 0u, 0u };     // groups 2/3: unused (2D tile)
      asm volatile("tensor_load_to_lds %0, %1, %2, %3"
                   :: "s"(g0), "s"(g1), "s"(gz), "s"(gz)
                   : "memory");
      tcs[lane] = t_col[bh * N_ + kk + lane];
      __builtin_amdgcn_s_wait_tensorcnt(0);
    }
    __syncthreads();   // TDM data + tcs visible to all waves

    if (kk + 32 < N_)
      __builtin_prefetch(sb + (size_t)gr * N_ + kk + 32, 0, 1);

    const __bf16* sp = sb + (size_t)gr * N_ + kk + kb;
    v8bf slo = *(const v8bf*)(sp);
    v8bf shi = *(const v8bf*)(sp + 16);
    v16bf pf;
#pragma unroll
    for (int j = 0; j < 8; ++j) {
      float x = bf2f(slo[j]);
      float p = (x >= trow || x >= tcs[kb + j]) ? __expf(x - mrow) : 0.f;
      rowsum += p; pf[j] = f2bf(p);
      float y = bf2f(shi[j]);
      float q = (y >= trow || y >= tcs[kb + 16 + j]) ? __expf(y - mrow) : 0.f;
      rowsum += q; pf[j + 8] = f2bf(q);
    }
#pragma unroll
    for (int j = 0; j < 4; ++j) {
      v16bf b = ldfragB_kc(Vs + j * 16, lane, HD_);
      acc[j] = wmma_bf16(pf, b, acc[j]);
    }
    __syncthreads();   // all reads of Vs done before next TDM overwrite
  }

  float tot = rowsum + __shfl_xor(rowsum, 16);       // lanes l, l^16 share a row
  float inv[8];
#pragma unroll
  for (int r = 0; r < 8; ++r)
    inv[r] = 1.f / __shfl(tot, r + ((lane >> 4) << 3));

  float* ob = O + (size_t)bh * N_ * HD_;
#pragma unroll
  for (int j = 0; j < 4; ++j)
#pragma unroll
    for (int r = 0; r < 8; ++r) {
      int row = n0 + r + ((lane >> 4) << 3);
      int col = j * 16 + (lane & 15);
      ob[(size_t)row * HD_ + col] = acc[j][r] * inv[r];
    }
}

// ---------------------------------------------------------------------------
// Kernel 5: out = (head-merged O) @ Wp^T + bp, fp32 output.
// ---------------------------------------------------------------------------
__global__ __launch_bounds__(256) void final_kernel(const float* __restrict__ O,
                                                    const float* __restrict__ Wp,
                                                    const float* __restrict__ bp,
                                                    float* __restrict__ out) {
  __shared__ __bf16 As[64 * 32];
  __shared__ __bf16 Ws[64 * 32];
  const int tid = threadIdx.x, lane = tid & 31, w = tid >> 5;
  const int m0 = blockIdx.y * 64;
  const int c0 = blockIdx.x * 64;
  const int sr = w >> 1;
  const int sc = (w & 1) * 2;

  v8f acc0 = {}, acc1 = {};
  for (int kk = 0; kk < C_; kk += 32) {
    for (int i = tid; i < 64 * 32; i += 256) {
      int r = i >> 5, c = i & 31;
      int m = m0 + r, k = kk + c;
      int b = m >> 11, n = m & (N_ - 1);
      int h = k >> 6, d = k & 63;
      As[i] = f2bf(O[(((size_t)(b * H_ + h) * N_ + n) << 6) + d]);
      Ws[i] = f2bf(Wp[(size_t)(c0 + r) * C_ + k]);
    }
    __syncthreads();
    v16bf a  = ldfragA(As + sr * 16 * 32, lane, 32);
    v16bf b0 = ldfragB(Ws + (sc * 16) * 32, lane, 32);
    v16bf b1 = ldfragB(Ws + ((sc + 1) * 16) * 32, lane, 32);
    acc0 = wmma_bf16(a, b0, acc0);
    acc1 = wmma_bf16(a, b1, acc1);
    __syncthreads();
  }
#pragma unroll
  for (int r = 0; r < 8; ++r) {
    int m = m0 + sr * 16 + r + ((lane >> 4) << 3);
#pragma unroll
    for (int j = 0; j < 2; ++j) {
      int c = c0 + (sc + j) * 16 + (lane & 15);
      float v = ((j == 0) ? acc0[r] : acc1[r]) + bp[c];
      out[(size_t)m * C_ + c] = v;
    }
  }
}

// ---------------------------------------------------------------------------
extern "C" void kernel_launch(void* const* d_in, const int* in_sizes, int n_in,
                              void* d_out, int out_size, void* d_ws, size_t ws_size,
                              hipStream_t stream) {
  const float* q  = (const float*)d_in[0];
  const float* kv = (const float*)d_in[1];
  const float* Wq = (const float*)d_in[2];
  const float* Wk = (const float*)d_in[3];
  const float* Wv = (const float*)d_in[4];
  const float* Wp = (const float*)d_in[5];
  const float* bp = (const float*)d_in[6];
  float* out = (float*)d_out;

  // Workspace carve-up (~148.4 MB total; all offsets 256B-aligned)
  char* ws = (char*)d_ws;
  const size_t nHead = (size_t)B_ * H_ * N_ * HD_;       // 2M elems
  __bf16* Qh = (__bf16*)ws;                  ws += nHead * 2;            // 4 MB
  __bf16* Kh = (__bf16*)ws;                  ws += nHead * 2;            // 4 MB
  __bf16* Vh = (__bf16*)ws;                  ws += nHead * 2;            // 4 MB
  __bf16* S  = (__bf16*)ws;                  ws += (size_t)BH_ * N_ * N_ * 2; // 128 MB
  float* trow = (float*)ws;                  ws += (size_t)BH_ * N_ * 4; // 128 KB
  float* mrow = (float*)ws;                  ws += (size_t)BH_ * N_ * 4;
  float* tcol = (float*)ws;                  ws += (size_t)BH_ * N_ * 4;
  float* O    = (float*)ws;                  ws += nHead * 4;            // 8 MB

  dim3 gProj(C_ / 64, (B_ * N_) / 64);
  proj_kernel<<<gProj, 256, 0, stream>>>(q,  Wq, Qh);
  proj_kernel<<<gProj, 256, 0, stream>>>(kv, Wk, Kh);
  proj_kernel<<<gProj, 256, 0, stream>>>(kv, Wv, Vh);

  dim3 gS(N_ / 128, N_ / 128, BH_);
  scores_kernel<<<gS, 256, 0, stream>>>(Qh, Kh, S);

  select_kernel<<<BH_ * N_, 256, 0, stream>>>(S, trow, mrow, 0);
  select_kernel<<<BH_ * N_, 256, 0, stream>>>(S, tcol, nullptr, 1);

  softmax_av_kernel<<<(BH_ * N_) / 128, 256, 0, stream>>>(S, Vh, trow, tcol,
                                                          mrow, O);

  final_kernel<<<gProj, 256, 0, stream>>>(O, Wp, bp, out);
}
